// ScaledDotProductAttention_32753420599920
// MI455X (gfx1250) — matrix operand order repaired and checked by reading, not compiled, not run on hardware
//
#include <hip/hip_runtime.h>
#include <math.h>

typedef __attribute__((ext_vector_type(16))) __bf16 v16bf;
typedef __attribute__((ext_vector_type(8)))  float  v8f;

#define S_LEN 2048
#define DKDIM 128
#define NHEAD 16
#define NBAT  2
#define QT    16              // queries per workgroup
#define SP    (S_LEN + 4)     // padded LDS row stride (floats)
#define SCALE 0.08838834764831845f   // 1/sqrt(128)

__device__ __forceinline__ __bf16 tobf(float x) { return (__bf16)x; }

__global__ __launch_bounds__(256) void attn_fwd_kernel(
    const float* __restrict__ Q, const float* __restrict__ K,
    const float* __restrict__ V, const int* __restrict__ M,
    float* __restrict__ Out, float* __restrict__ Attn)
{
  extern __shared__ float smem[];
  float* sc     = smem;               // [QT][SP] scores / probabilities
  float* red    = smem + QT * SP;     // [16][16] reduction scratch
  float* rowmx  = red + 256;          // [16]
  float* rowsum = rowmx + 16;         // [16]
  float* invs   = rowsum + 16;        // [16]

  const int bh    = blockIdx.y;             // b*H + h
  const int b     = bh / NHEAD;
  const int qbase = blockIdx.x * QT;
  const int tid   = threadIdx.x;
  const int wave  = tid >> 5;               // 0..7
  const int lane  = tid & 31;
  const int hi    = lane >> 4;              // half-wave
  const int ln    = lane & 15;

  const float* Qp = Q + (size_t)bh * S_LEN * DKDIM;
  const float* Kp = K + (size_t)bh * S_LEN * DKDIM;
  const float* Vp = V + (size_t)bh * S_LEN * DKDIM;
  const int*   Mp = M + (size_t)b  * S_LEN * S_LEN;
  float* Op = Out  + (size_t)bh * S_LEN * DKDIM;
  float* Ap = Attn + (size_t)bh * S_LEN * S_LEN;

  // ---- Q tile A-fragments (bf16), built once, reused for all 128 key tiles ----
  v16bf qa[4];
  #pragma unroll
  for (int ks = 0; ks < 4; ++ks) {
    const float* qrow = Qp + (size_t)(qbase + ln) * DKDIM + ks * 32;
    #pragma unroll
    for (int e = 0; e < 8; ++e) {
      qa[ks][e]     = tobf(qrow[8 * hi + e]);        // K = 8*hi + e
      qa[ks][e + 8] = tobf(qrow[16 + 8 * hi + e]);   // K = 16 + 8*hi + e
    }
  }

  // ---- scores = Q K^T * scale, masked; wave w handles key tiles w, w+8, ... ----
  for (int t = wave; t < S_LEN / 16; t += 8) {
    const int nb = t * 16;
    if (t + 8 < S_LEN / 16)  // prefetch next K tile rows (global_prefetch_b8)
      __builtin_prefetch(Kp + (size_t)(nb + 128 + ln) * DKDIM, 0, 1);
    v8f c = {};
    #pragma unroll
    for (int ks = 0; ks < 4; ++ks) {
      const float* krow = Kp + (size_t)(nb + ln) * DKDIM + ks * 32;
      v16bf bb;
      #pragma unroll
      for (int e = 0; e < 8; ++e) {
        bb[e]     = tobf(krow[8 * hi + e]);        // B[K=8*hi+e][N=ln]
        bb[e + 8] = tobf(krow[16 + 8 * hi + e]);   // B[K=16+8*hi+e][N=ln]
      }
      c = __builtin_amdgcn_wmma_f32_16x16x32_bf16(false, qa[ks], false, bb,
                                                  (short)0, c, false, false);
    }
    #pragma unroll
    for (int e = 0; e < 8; ++e) {
      const int m = e + 8 * hi;                     // row within query tile
      float s = c[e] * SCALE;
      const int mv = Mp[(size_t)(qbase + m) * S_LEN + nb + ln];
      s = (mv == 0) ? -1e-9f : s;                   // faithful: -1e-9, not -inf
      sc[m * SP + nb + ln] = s;
    }
  }
  __syncthreads();

  // ---- softmax (f32): 16 threads per row ----
  const int row = tid >> 4;
  const int sl  = tid & 15;
  float mx = -INFINITY;
  for (int j = 0; j < S_LEN / 16; ++j)
    mx = fmaxf(mx, sc[row * SP + sl + 16 * j]);
  red[row * 16 + sl] = mx;
  __syncthreads();
  if (sl == 0) {
    float m2 = red[row * 16];
    #pragma unroll
    for (int j = 1; j < 16; ++j) m2 = fmaxf(m2, red[row * 16 + j]);
    rowmx[row] = m2;
  }
  __syncthreads();
  const float rm = rowmx[row];
  float sum = 0.f;
  for (int j = 0; j < S_LEN / 16; ++j) {
    const int cc = sl + 16 * j;
    float p = __expf(sc[row * SP + cc] - rm);
    sc[row * SP + cc] = p;
    sum += p;
  }
  red[row * 16 + sl] = sum;
  __syncthreads();
  if (sl == 0) {
    float s2 = 0.f;
    #pragma unroll
    for (int j = 0; j < 16; ++j) s2 += red[row * 16 + j];
    rowsum[row] = s2;
    invs[row]   = 1.0f / s2;
  }
  __syncthreads();

  // ---- normalize in LDS; stream attn to HBM with non-temporal stores ----
  // (attn = 512 MB >> 192 MB L2; NT keeps K/V/mask working set resident)
  for (int i = tid; i < QT * S_LEN; i += 256) {
    const int r    = i >> 11;          // / 2048
    const int cidx = i & (S_LEN - 1);
    float p = sc[r * SP + cidx] * invs[r];
    sc[r * SP + cidx] = p;
    __builtin_nontemporal_store(p, &Ap[(size_t)(qbase + r) * S_LEN + cidx]);
  }
  __syncthreads();

  // ---- out = P @ V : wave w owns output columns [16w, 16w+16) ----
  const int nb2 = wave * 16;
  v8f o = {};
  for (int kt = 0; kt < S_LEN / 32; ++kt) {
    const int kb = kt * 32;
    v16bf pa, vb;
    #pragma unroll
    for (int e = 0; e < 8; ++e) {
      pa[e]     = tobf(sc[ln * SP + kb + 8 * hi + e]);
      pa[e + 8] = tobf(sc[ln * SP + kb + 16 + 8 * hi + e]);
    }
    #pragma unroll
    for (int e = 0; e < 16; ++e)   // lanes ln=0..15 read 16 consecutive floats
      vb[e] = tobf(Vp[(size_t)(kb + (e < 8 ? 8 * hi + e : 16 + 8 * hi + (e - 8))) * DKDIM + nb2 + ln]);
    o = __builtin_amdgcn_wmma_f32_16x16x32_bf16(false, pa, false, vb,
                                                (short)0, o, false, false);
  }
  #pragma unroll
  for (int e = 0; e < 8; ++e) {
    const int m = e + 8 * hi;
    __builtin_nontemporal_store(o[e], &Op[(size_t)(qbase + m) * DKDIM + nb2 + ln]);
  }
}

extern "C" void kernel_launch(void* const* d_in, const int* in_sizes, int n_in,
                              void* d_out, int out_size, void* d_ws, size_t ws_size,
                              hipStream_t stream) {
  const float* Q = (const float*)d_in[0];
  const float* K = (const float*)d_in[1];
  const float* V = (const float*)d_in[2];
  const int*   M = (const int*)d_in[3];
  float* Out  = (float*)d_out;
  float* Attn = (float*)d_out + (size_t)NBAT * NHEAD * S_LEN * DKDIM;

  const size_t shmem = (size_t)(QT * SP + 256 + 48) * sizeof(float);  // ~132 KB
  hipFuncSetAttribute((const void*)attn_fwd_kernel,
                      hipFuncAttributeMaxDynamicSharedMemorySize, (int)shmem);
  dim3 grid(S_LEN / QT, NBAT * NHEAD);
  attn_fwd_kernel<<<grid, 256, shmem, stream>>>(Q, K, V, M, Out, Attn);
}
